// LinearQwen3OmniMoeTalkerTextSparseMoeBlock_34041910788785
// MI455X (gfx1250) — matrix operand order repaired and checked by reading, not compiled, not run on hardware
//
#include <hip/hip_runtime.h>

// ---------------------------------------------------------------------------
// Qwen3-Omni MoE talker block for MI455X (gfx1250, wave32, WMMA bf16).
//   out = routed(top2 of 8 experts) + sigmoid(x@sgw) * shared_mlp(x)
// bf16 WMMA (16x16x32) GEMMs with fp32 accumulate, token gather by rowmap
// indirection (2/8 of dense expert FLOPs), fused silu(g)*u epilogues,
// double-buffered LDS staging, async global->LDS for bf16 A tiles.
// ---------------------------------------------------------------------------

#define T_TOKENS 8192
#define HID      2048
#define EXPERTS  8
#define IDIM     1408
#define SIDIM    5632

#define A_ASYNC  1   // use global_load_async_to_lds_b128 for bf16 A tiles

typedef __bf16 bf16;
typedef __attribute__((ext_vector_type(16))) __bf16 v16bf;
typedef __attribute__((ext_vector_type(8)))  __bf16 v8bf;
typedef __attribute__((ext_vector_type(4)))  __bf16 v4bf;
typedef __attribute__((ext_vector_type(8)))  float  v8f;

union BFrag { v16bf v; v8bf h[2]; };

__device__ __forceinline__ float warp_sum32(float s) {
#pragma unroll
  for (int off = 16; off > 0; off >>= 1) s += __shfl_xor(s, off, 32);
  return s;
}

__device__ __forceinline__ unsigned pack_bf16x2(float a, float b) {
  union { unsigned u; bf16 h[2]; } p;
  p.h[0] = (bf16)a;   // lower address / even k
  p.h[1] = (bf16)b;   // odd k
  return p.u;
}

// ---------------------------------------------------------------------------
// Router: one wave per token. logits -> softmax -> top2 -> renorm, plus the
// shared-expert gate scalar sigmoid(x . sgw).
// ---------------------------------------------------------------------------
__global__ __launch_bounds__(256) void router_kernel(
    const float* __restrict__ x, const float* __restrict__ rw,
    const float* __restrict__ sgw, float* __restrict__ gate,
    int* __restrict__ topi, float* __restrict__ topw, int* __restrict__ cnt)
{
  const int t = blockIdx.x * 8 + (threadIdx.x >> 5);
  const int lane = threadIdx.x & 31;
  const float* xr = x + (size_t)t * HID;

  float logits[EXPERTS];
#pragma unroll
  for (int e = 0; e < EXPERTS; ++e) {
    const float* w = rw + (size_t)e * HID;
    float s = 0.f;
    for (int h = lane; h < HID; h += 32) s += xr[h] * w[h];
    logits[e] = warp_sum32(s);
  }
  float gs = 0.f;
  for (int h = lane; h < HID; h += 32) gs += xr[h] * sgw[h];
  gs = warp_sum32(gs);

  if (lane == 0) {
    float mx = logits[0];
#pragma unroll
    for (int e = 1; e < EXPERTS; ++e) mx = fmaxf(mx, logits[e]);
    float p[EXPERTS];
    float sum = 0.f;
#pragma unroll
    for (int e = 0; e < EXPERTS; ++e) { p[e] = __expf(logits[e] - mx); sum += p[e]; }
    const float inv = 1.f / sum;
#pragma unroll
    for (int e = 0; e < EXPERTS; ++e) p[e] *= inv;

    int i1 = 0;
#pragma unroll
    for (int e = 1; e < EXPERTS; ++e) if (p[e] > p[i1]) i1 = e;
    int i2 = (i1 == 0) ? 1 : 0;
#pragma unroll
    for (int e = 0; e < EXPERTS; ++e) if (e != i1 && p[e] > p[i2]) i2 = e;

    const float rs = 1.f / (p[i1] + p[i2]);
    topi[2 * t + 0] = i1;
    topi[2 * t + 1] = i2;
    topw[2 * t + 0] = p[i1] * rs;
    topw[2 * t + 1] = p[i2] * rs;
    gate[t] = 1.f / (1.f + __expf(-gs));
    atomicAdd(&cnt[i1], 1);
    atomicAdd(&cnt[i2], 1);
  }
}

__global__ void init_cnt_kernel(int* __restrict__ cnt) {
  if (threadIdx.x < EXPERTS) cnt[threadIdx.x] = 0;
}

__global__ void scan_kernel(const int* __restrict__ cnt,
                            int* __restrict__ off, int* __restrict__ cur) {
  if (threadIdx.x == 0 && blockIdx.x == 0) {
    int a = 0;
    for (int e = 0; e < EXPERTS; ++e) { off[e] = a; cur[e] = a; a += cnt[e]; }
    off[EXPERTS] = a;
  }
}

__global__ __launch_bounds__(256) void build_rows_kernel(
    const int* __restrict__ topi, const float* __restrict__ topw,
    int* __restrict__ cur, int* __restrict__ rowmap, float* __restrict__ roww)
{
  const int idx = blockIdx.x * 256 + threadIdx.x;
  if (idx >= T_TOKENS * 2) return;
  const int e = topi[idx];
  const int row = atomicAdd(&cur[e], 1);
  rowmap[row] = idx >> 1;
  roww[row] = topw[idx];
}

// out = gate[t] * S[t,h]   (initializes the whole output tensor)
__global__ __launch_bounds__(256) void combine_init_kernel(
    const float* __restrict__ S, const float* __restrict__ gate,
    float* __restrict__ out)
{
  const int idx = blockIdx.x * 256 + threadIdx.x;
  const int t = idx / HID;
  out[idx] = gate[t] * S[idx];
}

// ---------------------------------------------------------------------------
// WMMA GEMM, bf16 operands (f32 converted during staging), fp32 accumulate.
//   C[M,N] = A[M,K] x B[K,N]
// Tile BM=128, BN = DUAL?64:128, BK=32; 256 threads = 8 waves in 4(M)x2(N);
// wave tile 32 x BN/2 via v_wmma_f32_16x16x32_bf16. Double-buffered LDS.
//   DUAL    : two B matrices, epilogue writes bf16 silu(g)*u (*row weight)
//   A_IND   : A row indices from rowmap (token gather by indirection)
//   SCATTER : output rows scattered to out[rowmap[row]] with +=
// Out-of-range rows are CLAMPED (not masked) so staging loads are branch-free
// and the rowmap lookups hoist out of the K loop; clamped rows produce
// garbage accumulators that the epilogue never stores.
// EXEC is uniform wherever WMMA executes (ISA 7.12 requirement).
// ---------------------------------------------------------------------------
template<bool A_BF16, bool DUAL, bool A_IND, bool SCATTER>
__global__ __launch_bounds__(256) void wmma_gemm(
    const void* __restrict__ Aptr, const float* __restrict__ B1,
    const float* __restrict__ B2, void* __restrict__ Out,
    int N, int K, int lda, int ldo, int M_fixed,
    const int* __restrict__ seg_off, int seg_idx,
    const int* __restrict__ rowmap, const float* __restrict__ roww)
{
  constexpr int BM    = 128;
  constexpr int BN    = DUAL ? 64 : 128;
  constexpr int NMAT  = DUAL ? 2 : 1;
  constexpr int NSUB  = BN / 32;          // 16-wide N subtiles per wave
  constexpr int APASS = A_BF16 ? 2 : 4;   // staging passes over A rows
  constexpr int RSTEP = A_BF16 ? 64 : 32;
  constexpr int BPASS = DUAL ? 1 : 2;     // staging passes over B k-pairs

  int m0 = 0, M_eff = M_fixed;
  if (seg_off) {                          // per-expert segment GEMM
    m0 = seg_off[seg_idx];
    M_eff = seg_off[seg_idx + 1] - m0;
  }
  const int bm = blockIdx.y * BM;
  if (bm >= M_eff) return;                // block-uniform early exit
  const int bn = blockIdx.x * BN;

  __shared__ __align__(16) bf16 As[2][BM][32];
  __shared__ __align__(16) bf16 Bs[2][NMAT][BN][32];  // transposed: [n][k]

  const int tid   = threadIdx.x;
  const int wid   = tid >> 5;
  const int lane  = tid & 31;
  const int wm    = (wid & 3) * 32;         // wave M base within tile
  const int wn    = (wid >> 2) * (BN / 2);  // wave N base within tile
  const int l16   = lane & 15;
  const int khalf = lane >> 4;

  // ---- per-thread staging coordinates (fixed across K) ----
  int rA, kA;
  if constexpr (A_BF16) { rA = tid >> 2; kA = (tid & 3) * 8; }
  else                  { rA = tid >> 3; kA = (tid & 7) * 4; }
  int k0B, n4B;
  if constexpr (DUAL) { k0B = (tid >> 4) * 2; n4B = (tid & 15) * 4; }
  else                { k0B = (tid >> 5) * 2; n4B = (tid & 31) * 4; }

  // A row base pointers, clamped + (optionally) rowmap-indirected, hoisted.
  const void* aptr[APASS];
#pragma unroll
  for (int p = 0; p < APASS; ++p) {
    const int rg = bm + rA + p * RSTEP;
    const int rc = rg < M_eff ? rg : M_eff - 1;
    int row;
    if constexpr (A_IND) row = rowmap[m0 + rc]; else row = m0 + rc;
    if constexpr (A_BF16)
      aptr[p] = (const void*)((const bf16*)Aptr + (size_t)row * lda + kA);
    else
      aptr[p] = (const void*)((const float*)Aptr + (size_t)row * lda + kA);
  }

  v8f acc[NMAT][2][NSUB];
#pragma unroll
  for (int mat = 0; mat < NMAT; ++mat)
#pragma unroll
    for (int ms = 0; ms < 2; ++ms)
#pragma unroll
      for (int ns = 0; ns < NSUB; ++ns)
#pragma unroll
        for (int i = 0; i < 8; ++i) acc[mat][ms][ns][i] = 0.f;

  // staging registers (live across the WMMA block for overlap)
  float4 aF[4];
  v8bf   aH[2];
  float4 bR[NMAT][BPASS][2];

  // ---- issue global loads for tile kt (async A goes straight to LDS) ----
  auto load_tile = [&](int kt, int nbuf) {
    if constexpr (A_BF16) {
#if A_ASYNC
#pragma unroll
      for (int p = 0; p < APASS; ++p) {
        // LDS aperture keeps the byte offset in the low 32 bits.
        unsigned lds_a = (unsigned)(uintptr_t)&As[nbuf][rA + p * RSTEP][kA];
        unsigned long long ga =
            (unsigned long long)(uintptr_t)((const bf16*)aptr[p] + kt);
        asm volatile("global_load_async_to_lds_b128 %0, %1, off"
                     :: "v"(lds_a), "v"(ga) : "memory");
      }
#else
#pragma unroll
      for (int p = 0; p < APASS; ++p)
        aH[p] = *(const v8bf*)((const bf16*)aptr[p] + kt);
#endif
    } else {
#pragma unroll
      for (int p = 0; p < APASS; ++p)
        aF[p] = *(const float4*)((const float*)aptr[p] + kt);
    }
#pragma unroll
    for (int mat = 0; mat < NMAT; ++mat) {
      const float* Bp = mat ? B2 : B1;
#pragma unroll
      for (int p = 0; p < BPASS; ++p) {
        const float* base =
            Bp + (size_t)(kt + k0B + p * 16) * N + bn + n4B;
        bR[mat][p][0] = *(const float4*)base;
        bR[mat][p][1] = *(const float4*)(base + N);
      }
    }
    (void)nbuf;
  };

  // ---- convert + store staged registers into LDS buffer nbuf ----
  auto store_tile = [&](int nbuf) {
    if constexpr (A_BF16) {
#if !A_ASYNC
#pragma unroll
      for (int p = 0; p < APASS; ++p)
        *(v8bf*)&As[nbuf][rA + p * RSTEP][kA] = aH[p];
#endif
    } else {
#pragma unroll
      for (int p = 0; p < APASS; ++p) {
        v4bf v;
        v[0] = (bf16)aF[p].x; v[1] = (bf16)aF[p].y;
        v[2] = (bf16)aF[p].z; v[3] = (bf16)aF[p].w;
        *(v4bf*)&As[nbuf][rA + p * RSTEP][kA] = v;
      }
    }
#pragma unroll
    for (int mat = 0; mat < NMAT; ++mat) {
#pragma unroll
      for (int p = 0; p < BPASS; ++p) {
        const int k0 = k0B + p * 16;
        const float c0[4] = {bR[mat][p][0].x, bR[mat][p][0].y,
                             bR[mat][p][0].z, bR[mat][p][0].w};
        const float c1[4] = {bR[mat][p][1].x, bR[mat][p][1].y,
                             bR[mat][p][1].z, bR[mat][p][1].w};
#pragma unroll
        for (int j = 0; j < 4; ++j)
          *(unsigned*)&Bs[nbuf][mat][n4B + j][k0] = pack_bf16x2(c0[j], c1[j]);
      }
    }
  };

  // ---- WMMA over LDS buffer cbuf ----
  auto compute = [&](int cbuf) {
    // A 16x32 bf16 frag: lanes 0-15 row M=l16, K={0-7,16-23};
    //                    lanes 16-31 same rows, K={8-15,24-31}.
    BFrag a[2];
#pragma unroll
    for (int ms = 0; ms < 2; ++ms) {
      const v8bf* ap = (const v8bf*)&As[cbuf][wm + ms * 16 + l16][0];
      a[ms].h[0] = ap[khalf];
      a[ms].h[1] = ap[khalf + 2];
    }
#pragma unroll
    for (int mat = 0; mat < NMAT; ++mat) {
#pragma unroll
      for (int ns = 0; ns < NSUB; ++ns) {
        // B 32x16 frag: lanes 0-15 col N=l16 K={0-7,16-23}; lanes 16-31 K={8-15,24-31}.
        BFrag b;
        const v8bf* bp = (const v8bf*)&Bs[cbuf][mat][wn + ns * 16 + l16][0];
        b.h[0] = bp[khalf];
        b.h[1] = bp[khalf + 2];
#pragma unroll
        for (int ms = 0; ms < 2; ++ms) {
          acc[mat][ms][ns] = __builtin_amdgcn_wmma_f32_16x16x32_bf16(
              false, a[ms].v, false, b.v, (short)0, acc[mat][ms][ns],
              false, false);
        }
      }
    }
  };

  // ---- pipelined main loop (double-buffered LDS) ----
  load_tile(0, 0);
  store_tile(0);
  if constexpr (A_BF16) {
#if A_ASYNC
    asm volatile("s_wait_asynccnt 0x0" ::: "memory");
#endif
  }
  __syncthreads();

  int buf = 0;
  for (int kt = 32; kt < K; kt += 32) {
    load_tile(kt, buf ^ 1);     // prefetch next tile (overlaps WMMA below)
    compute(buf);
    store_tile(buf ^ 1);
    if constexpr (A_BF16) {
#if A_ASYNC
      asm volatile("s_wait_asynccnt 0x0" ::: "memory");
#endif
    }
    __syncthreads();
    buf ^= 1;
  }
  compute(buf);

  // ---- epilogue ----
  // C/D layout: lane holds col N=l16; VGPR v -> row v (lanes 0-15) or 8+v
  // (lanes 16-31) within each 16-row subtile.
#pragma unroll
  for (int ms = 0; ms < 2; ++ms) {
#pragma unroll
    for (int ns = 0; ns < NSUB; ++ns) {
      const int col = bn + wn + ns * 16 + l16;
#pragma unroll
      for (int v = 0; v < 8; ++v) {
        const int rloc = wm + ms * 16 + khalf * 8 + v;
        const int rg = bm + rloc;
        if (rg >= M_eff) continue;
        if constexpr (DUAL) {
          const float g = acc[0][ms][ns][v];
          const float u = acc[1][ms][ns][v];
          float val = (g / (1.f + __expf(-g))) * u;     // silu(g) * u
          if (roww) val *= roww[m0 + rg];               // fold routing weight
          ((bf16*)Out)[(size_t)(m0 + rg) * ldo + col] = (bf16)val;
        } else {
          const float val = acc[0][ms][ns][v];
          if constexpr (SCATTER) {
            const int t = rowmap[m0 + rg];
            // Safe non-atomic +=: one row per token per expert; expert
            // kernels are stream-serialized.
            ((float*)Out)[(size_t)t * ldo + col] += val;
          } else {
            ((float*)Out)[(size_t)(m0 + rg) * ldo + col] = val;
          }
        }
      }
    }
  }
}

// ---------------------------------------------------------------------------
extern "C" void kernel_launch(void* const* d_in, const int* in_sizes, int n_in,
                              void* d_out, int out_size, void* d_ws, size_t ws_size,
                              hipStream_t stream)
{
  const float* x   = (const float*)d_in[0];  // [T, H]
  const float* rw  = (const float*)d_in[1];  // [E, H]
  const float* wg  = (const float*)d_in[2];  // [E, H, I]
  const float* wu  = (const float*)d_in[3];  // [E, H, I]
  const float* wd  = (const float*)d_in[4];  // [E, I, H]
  const float* swg = (const float*)d_in[5];  // [H, SI]
  const float* swu = (const float*)d_in[6];  // [H, SI]
  const float* swd = (const float*)d_in[7];  // [SI, H]
  const float* sgw = (const float*)d_in[8];  // [H]
  float* out = (float*)d_out;                // [T, H]

  char* p = (char*)d_ws;
  auto alloc = [&](size_t bytes) -> char* {
    char* r = p;
    p += (bytes + 255) & ~(size_t)255;
    return r;
  };
  float* gate   = (float*)alloc((size_t)T_TOKENS * sizeof(float));
  int*   topi   = (int*)  alloc((size_t)T_TOKENS * 2 * sizeof(int));
  float* topw   = (float*)alloc((size_t)T_TOKENS * 2 * sizeof(float));
  int*   cnt    = (int*)  alloc(EXPERTS * sizeof(int));
  int*   off    = (int*)  alloc((EXPERTS + 1) * sizeof(int));
  int*   cur    = (int*)  alloc(EXPERTS * sizeof(int));
  int*   rowmap = (int*)  alloc((size_t)T_TOKENS * 2 * sizeof(int));
  float* roww   = (float*)alloc((size_t)T_TOKENS * 2 * sizeof(float));
  float* S      = (float*)alloc((size_t)T_TOKENS * HID * sizeof(float));
  bf16*  act    = (bf16*) alloc((size_t)T_TOKENS * SIDIM * sizeof(bf16));

  // 1) routing
  init_cnt_kernel<<<1, 32, 0, stream>>>(cnt);
  router_kernel<<<T_TOKENS / 8, 256, 0, stream>>>(x, rw, sgw, gate, topi, topw, cnt);
  scan_kernel<<<1, 1, 0, stream>>>(cnt, off, cur);
  build_rows_kernel<<<(T_TOKENS * 2) / 256, 256, 0, stream>>>(topi, topw, cur, rowmap, roww);

  // 2) shared expert: act = silu(x@swg) * (x@swu); S = act @ swd
  wmma_gemm<false, true, false, false>
      <<<dim3(SIDIM / 64, T_TOKENS / 128), 256, 0, stream>>>(
          x, swg, swu, act, SIDIM, HID, HID, SIDIM, T_TOKENS,
          nullptr, 0, nullptr, nullptr);
  wmma_gemm<true, false, false, false>
      <<<dim3(HID / 128, T_TOKENS / 128), 256, 0, stream>>>(
          act, swd, nullptr, S, HID, SIDIM, SIDIM, HID, T_TOKENS,
          nullptr, 0, nullptr, nullptr);

  // 3) out = sigmoid(x.sgw) * S
  combine_init_kernel<<<(T_TOKENS * HID) / 256, 256, 0, stream>>>(S, gate, out);

  // 4) routed experts on gathered segments (indirect A rows); routing weight
  //    folded into the bf16 activation.
  for (int e = 0; e < EXPERTS; ++e) {
    wmma_gemm<false, true, true, false>
        <<<dim3(IDIM / 64, T_TOKENS / 128), 256, 0, stream>>>(
            x, wg + (size_t)e * HID * IDIM, wu + (size_t)e * HID * IDIM,
            act, IDIM, HID, HID, IDIM, 0, off, e, rowmap, roww);
  }
  // 5) down-projection, scattered += into out
  for (int e = 0; e < EXPERTS; ++e) {
    wmma_gemm<true, false, false, true>
        <<<dim3(HID / 128, T_TOKENS / 128), 256, 0, stream>>>(
            act, wd + (size_t)e * IDIM * HID, nullptr, out,
            HID, IDIM, IDIM, HID, 0, off, e, rowmap, nullptr);
  }
  (void)in_sizes; (void)n_in; (void)out_size; (void)ws_size;
}
